// PCN_decoder_68487548502104
// MI455X (gfx1250) — matrix operand order repaired and checked by reading, not compiled, not run on hardware
//
#include <hip/hip_runtime.h>

// ---------------------------------------------------------------------------
// MI455X (gfx1250) PCN decoder.
// All GEMMs use v_wmma_f32_16x16x32_bf16 (bf16 in / f32 acc).
// Data-movement decisions (23.3 TB/s HBM, 192 MB L2):
//  * edge MLP (3->64->128->128 + max over K): fused, activations bf16 in LDS.
//  * final conv layer 1 (2053->512): 269 MB "feat" concat gathered on the fly.
//  * gemm_cm exploits uniform C%32==0 / P%128==0 so tile loaders have no
//    per-lane guards (no exec-mask juggling); fast/slow choices are uniform.
// ---------------------------------------------------------------------------

typedef __attribute__((ext_vector_type(16))) __bf16 v16bf;
typedef __attribute__((ext_vector_type(8)))  float  v8f;
typedef __attribute__((ext_vector_type(2)))  unsigned long long ull2;

__device__ __forceinline__ __bf16 f2bf(float f) {
  union { float f; unsigned u; } a; a.f = f;
  unsigned r = a.u + 0x7FFFu + ((a.u >> 16) & 1u);   // round-to-nearest-even
  unsigned short h = (unsigned short)(r >> 16);
  __bf16 o; __builtin_memcpy(&o, &h, 2); return o;
}

__device__ __forceinline__ v8f v8f_zero() {
  v8f z;
#pragma unroll
  for (int i = 0; i < 8; ++i) z[i] = 0.f;
  return z;
}

union bf16x16 { v16bf v; ull2 q[2]; __bf16 e[16]; };

__device__ __forceinline__ bf16x16 cvt16(float4 f0, float4 f1, float4 f2, float4 f3) {
  bf16x16 u;
  u.e[0]=f2bf(f0.x); u.e[1]=f2bf(f0.y); u.e[2]=f2bf(f0.z); u.e[3]=f2bf(f0.w);
  u.e[4]=f2bf(f1.x); u.e[5]=f2bf(f1.y); u.e[6]=f2bf(f1.z); u.e[7]=f2bf(f1.w);
  u.e[8]=f2bf(f2.x); u.e[9]=f2bf(f2.y); u.e[10]=f2bf(f2.z); u.e[11]=f2bf(f2.w);
  u.e[12]=f2bf(f3.x); u.e[13]=f2bf(f3.y); u.e[14]=f2bf(f3.z); u.e[15]=f2bf(f3.w);
  return u;
}

// A fragment (16x32 bf16): lane l -> M = l%16 ; elems 0-7 -> K = (l/16)*8 + i,
// elems 8-15 -> K = (l/16)*8 + 16 + (i-8).  Two 16B LDS reads.
__device__ __forceinline__ v16bf lds_frag_a(const __bf16* row, int kb) {
  union { v16bf v; ull2 q[2]; } u;
  u.q[0] = *(const ull2*)(row + kb);
  u.q[1] = *(const ull2*)(row + kb + 16);
  return u.v;
}
// B fragment (32x16 bf16): lane l -> N = l%16 ; elems 0-7 -> K = (l/16)*8 + i,
// elems 8-15 -> K = (l/16)*8 + 16 + (i-8).
__device__ __forceinline__ v16bf lds_frag_b(const __bf16* row, int kb2) {
  union { v16bf v; ull2 q[2]; } u;
  u.q[0] = *(const ull2*)(row + kb2);
  u.q[1] = *(const ull2*)(row + kb2 + 16);
  return u.v;
}

// weight fragment straight from global: elems 0-7 at wr[0..7], 8-15 at wr[16..23]
__device__ __forceinline__ v16bf wfrag_global(const float* __restrict__ wr) {
  float4 f0 = *(const float4*)(wr);
  float4 f1 = *(const float4*)(wr + 4);
  float4 f2 = *(const float4*)(wr + 16);
  float4 f3 = *(const float4*)(wr + 20);
  return cvt16(f0, f1, f2, f3).v;
}

// ---------------------------------------------------------------------------
// gemm_cm: Out[o][p] = act( W[o][:] . In[:][p] + b[o] ),  In/Out ld == P.
// PRECONDITIONS (uniform, guaranteed by call sites): C % 32 == 0, P % 128 == 0,
// P % 4 == 0, pointers 16B-aligned.  O arbitrary (guard only when needed).
// Block tile 128x128, 8 waves (4M x 2N), K chunked by 32.
// ---------------------------------------------------------------------------
__global__ __launch_bounds__(256) void gemm_cm(
    const float* __restrict__ W, const float* __restrict__ bias,
    const float* __restrict__ In, float* __restrict__ Out,
    int O, int C, int P, int relu)
{
  __shared__ __bf16 As[128][40];   // [o][k]
  __shared__ __bf16 Bs[128][40];   // [p][k]
  const int tid = threadIdx.x;
  const int w = tid >> 5, lane = tid & 31, lo = lane & 15, hi = lane >> 4;
  const int p0 = blockIdx.x * 128, o0 = blockIdx.y * 128;
  const int lrow = tid >> 1, lcol = (tid & 1) * 16;
  const bool fullO = (o0 + 128 <= O) && ((C & 3) == 0);   // uniform fast flag

  v8f acc[2][4];
#pragma unroll
  for (int mi = 0; mi < 2; ++mi)
#pragma unroll
    for (int ni = 0; ni < 4; ++ni) acc[mi][ni] = v8f_zero();

  for (int c0 = 0; c0 < C; c0 += 32) {
    // ---- A (weights, row-major [O][C]) ----
    {
      const float* src = W + (long long)(o0 + lrow) * C + c0 + lcol;
      if (fullO) {                       // uniform branch, no lane guards
        bf16x16 u = cvt16(*(const float4*)(src),     *(const float4*)(src + 4),
                          *(const float4*)(src + 8), *(const float4*)(src + 12));
        *(ull2*)&As[lrow][lcol]     = u.q[0];
        *(ull2*)&As[lrow][lcol + 8] = u.q[1];
        if (c0 + 32 < C) __builtin_prefetch(src + 32, 0, 0);   // global_prefetch_b8
      } else {
        int o = o0 + lrow;
#pragma unroll
        for (int i = 0; i < 16; ++i)
          As[lrow][lcol + i] = f2bf((o < O) ? src[i] : 0.f);
      }
    }
    // ---- B (activations [C][P], unit p stride): k-major, fully coalesced ----
    {
      int k = tid >> 3, pi0 = (tid & 7) * 16;
      const float* src = In + (long long)(c0 + k) * P + p0 + pi0;
      bf16x16 u = cvt16(*(const float4*)(src),     *(const float4*)(src + 4),
                        *(const float4*)(src + 8), *(const float4*)(src + 12));
#pragma unroll
      for (int i = 0; i < 16; ++i) Bs[pi0 + i][k] = u.e[i];
    }
    __syncthreads();

    v16bf af[2], bf[4];
#pragma unroll
    for (int mi = 0; mi < 2; ++mi)
      af[mi] = lds_frag_a(&As[(w >> 1) * 32 + mi * 16 + lo][0], hi * 8);
#pragma unroll
    for (int ni = 0; ni < 4; ++ni)
      bf[ni] = lds_frag_b(&Bs[(w & 1) * 64 + ni * 16 + lo][0], hi * 8);
#pragma unroll
    for (int mi = 0; mi < 2; ++mi)
#pragma unroll
      for (int ni = 0; ni < 4; ++ni)
        acc[mi][ni] = __builtin_amdgcn_wmma_f32_16x16x32_bf16(
            false, af[mi], false, bf[ni], (short)0, acc[mi][ni], false, false);
    __syncthreads();
  }

  // D layout: lane l, vgpr v -> N = l%16, M = v + (l/16)*8
  if (fullO) {
#pragma unroll
    for (int mi = 0; mi < 2; ++mi)
#pragma unroll
      for (int ni = 0; ni < 4; ++ni)
#pragma unroll
        for (int v = 0; v < 8; ++v) {
          int o = o0 + (w >> 1) * 32 + mi * 16 + v + hi * 8;
          int p = p0 + (w & 1) * 64 + ni * 16 + lo;
          float val = acc[mi][ni][v] + bias[o];
          if (relu) val = fmaxf(val, 0.f);
          Out[(long long)o * P + p] = val;
        }
  } else {
#pragma unroll
    for (int mi = 0; mi < 2; ++mi)
#pragma unroll
      for (int ni = 0; ni < 4; ++ni)
#pragma unroll
        for (int v = 0; v < 8; ++v) {
          int o = o0 + (w >> 1) * 32 + mi * 16 + v + hi * 8;
          int p = p0 + (w & 1) * 64 + ni * 16 + lo;
          if (o < O) {
            float val = acc[mi][ni][v] + bias[o];
            if (relu) val = fmaxf(val, 0.f);
            Out[(long long)o * P + p] = val;
          }
        }
  }
}

// ---------------------------------------------------------------------------
// gemm_rm: Out[o*sOutO + p*sOutP] = act( W[o][:] . In[p][:] + b[o] ).
// In row-major [P][ldInP], only used by the tiny fc layers (P == 8).
// PRECONDITIONS: C % 32 == 0, O % 128 == 0, ldInP % 4 == 0.
// ---------------------------------------------------------------------------
__global__ __launch_bounds__(256) void gemm_rm(
    const float* __restrict__ W, const float* __restrict__ bias,
    const float* __restrict__ In, long long ldInP,
    float* __restrict__ Out, long long sOutO, long long sOutP,
    int O, int C, int P, int relu)
{
  __shared__ __bf16 As[128][40];
  __shared__ __bf16 Bs[128][40];
  const int tid = threadIdx.x;
  const int w = tid >> 5, lane = tid & 31, lo = lane & 15, hi = lane >> 4;
  const int o0 = blockIdx.y * 128;
  const int lrow = tid >> 1, lcol = (tid & 1) * 16;

  v8f acc[2][4];
#pragma unroll
  for (int mi = 0; mi < 2; ++mi)
#pragma unroll
    for (int ni = 0; ni < 4; ++ni) acc[mi][ni] = v8f_zero();

  for (int c0 = 0; c0 < C; c0 += 32) {
    {   // A: O%128==0 and C%32==0 -> unguarded
      const float* src = W + (long long)(o0 + lrow) * C + c0 + lcol;
      bf16x16 u = cvt16(*(const float4*)(src),     *(const float4*)(src + 4),
                        *(const float4*)(src + 8), *(const float4*)(src + 12));
      *(ull2*)&As[lrow][lcol]     = u.q[0];
      *(ull2*)&As[lrow][lcol + 8] = u.q[1];
    }
    {   // B: row-major per position; zero-fill rows beyond P (uniform-ish branch)
      int p = lrow;
      if (p < P) {
        const float* src = In + (long long)p * ldInP + c0 + lcol;
        bf16x16 u = cvt16(*(const float4*)(src),     *(const float4*)(src + 4),
                          *(const float4*)(src + 8), *(const float4*)(src + 12));
        *(ull2*)&Bs[lrow][lcol]     = u.q[0];
        *(ull2*)&Bs[lrow][lcol + 8] = u.q[1];
      } else {
        ull2 z = {0ULL, 0ULL};
        *(ull2*)&Bs[lrow][lcol]     = z;
        *(ull2*)&Bs[lrow][lcol + 8] = z;
      }
    }
    __syncthreads();

    v16bf af[2], bf[4];
#pragma unroll
    for (int mi = 0; mi < 2; ++mi)
      af[mi] = lds_frag_a(&As[(w >> 1) * 32 + mi * 16 + lo][0], hi * 8);
#pragma unroll
    for (int ni = 0; ni < 4; ++ni)
      bf[ni] = lds_frag_b(&Bs[(w & 1) * 64 + ni * 16 + lo][0], hi * 8);
#pragma unroll
    for (int mi = 0; mi < 2; ++mi)
#pragma unroll
      for (int ni = 0; ni < 4; ++ni)
        acc[mi][ni] = __builtin_amdgcn_wmma_f32_16x16x32_bf16(
            false, af[mi], false, bf[ni], (short)0, acc[mi][ni], false, false);
    __syncthreads();
  }

#pragma unroll
  for (int mi = 0; mi < 2; ++mi)
#pragma unroll
    for (int ni = 0; ni < 4; ++ni)
#pragma unroll
      for (int v = 0; v < 8; ++v) {
        int o = o0 + (w >> 1) * 32 + mi * 16 + v + hi * 8;
        int p = (w & 1) * 64 + ni * 16 + lo;
        if (p < P) {
          float val = acc[mi][ni][v] + bias[o];
          if (relu) val = fmaxf(val, 0.f);
          Out[(long long)o * sOutO + (long long)p * sOutP] = val;
        }
      }
}

// ---------------------------------------------------------------------------
// Final-conv layer-1 GEMM with on-the-fly feat gather (2053 channels):
//   c<2: grid, c<5: sp coords, c<1029: merge features, else global x.
// ---------------------------------------------------------------------------
__device__ __forceinline__ float feat_val(int c, int p,
    const float* __restrict__ x, const float* __restrict__ sp,
    const float* __restrict__ lf) {
  int b = p >> 12, j = p & 4095, n = j >> 2, t = j & 3;
  if (c < 2) {
    float gx = (t < 2) ? -0.05f : 0.05f;
    float gy = (t & 1) ? 0.05f : -0.05f;
    return (c == 0) ? gx : gy;
  } else if (c < 5) {
    return sp[(b * 3 + (c - 2)) * 1024 + n];
  } else if (c < 1029) {
    return lf[(long long)(c - 5) * 8192 + b * 1024 + n];
  } else {
    return x[b * 1024 + (c - 1029)];
  }
}

__global__ __launch_bounds__(256) void gemm_featgather(
    const float* __restrict__ W, const float* __restrict__ bias,
    const float* __restrict__ x, const float* __restrict__ sp,
    const float* __restrict__ lf, float* __restrict__ Out, int O, int relu)
{
  const int C = 2053, P = 32768;
  __shared__ __bf16 As[128][40];
  __shared__ __bf16 Bs[128][40];
  const int tid = threadIdx.x;
  const int w = tid >> 5, lane = tid & 31, lo = lane & 15, hi = lane >> 4;
  const int p0 = blockIdx.x * 128, o0 = blockIdx.y * 128;
  const int lrow = tid >> 1, lcol = (tid & 1) * 16;

  v8f acc[2][4];
#pragma unroll
  for (int mi = 0; mi < 2; ++mi)
#pragma unroll
    for (int ni = 0; ni < 4; ++ni) acc[mi][ni] = v8f_zero();

  for (int c0 = 0; c0 < C; c0 += 32) {
    {   // A: rows are odd-aligned (C=2053) -> scalar copies; uniform guard only
      const float* src = W + (long long)(o0 + lrow) * C + c0 + lcol;
      if (o0 + 128 <= O && c0 + 32 <= C) {          // uniform fast flag
#pragma unroll
        for (int i = 0; i < 16; ++i) As[lrow][lcol + i] = f2bf(src[i]);
      } else {
        int o = o0 + lrow;
#pragma unroll
        for (int i = 0; i < 16; ++i) {
          int cc = c0 + lcol + i;
          As[lrow][lcol + i] = f2bf((o < O && cc < C) ? src[i] : 0.f);
        }
      }
    }
    {   // gathered B tile: k-major so the lf-region reads coalesce along p
      int k = tid >> 3, pi0 = (tid & 7) * 16;
      int cc = c0 + k;
#pragma unroll
      for (int i = 0; i < 16; ++i) {
        int p = p0 + pi0 + i;
        Bs[pi0 + i][k] = f2bf((cc < C) ? feat_val(cc, p, x, sp, lf) : 0.f);
      }
    }
    __syncthreads();

    v16bf af[2], bf[4];
#pragma unroll
    for (int mi = 0; mi < 2; ++mi)
      af[mi] = lds_frag_a(&As[(w >> 1) * 32 + mi * 16 + lo][0], hi * 8);
#pragma unroll
    for (int ni = 0; ni < 4; ++ni)
      bf[ni] = lds_frag_b(&Bs[(w & 1) * 64 + ni * 16 + lo][0], hi * 8);
#pragma unroll
    for (int mi = 0; mi < 2; ++mi)
#pragma unroll
      for (int ni = 0; ni < 4; ++ni)
        acc[mi][ni] = __builtin_amdgcn_wmma_f32_16x16x32_bf16(
            false, af[mi], false, bf[ni], (short)0, acc[mi][ni], false, false);
    __syncthreads();
  }

#pragma unroll
  for (int mi = 0; mi < 2; ++mi)
#pragma unroll
    for (int ni = 0; ni < 4; ++ni)
#pragma unroll
      for (int v = 0; v < 8; ++v) {
        int o = o0 + (w >> 1) * 32 + mi * 16 + v + hi * 8;
        int p = p0 + (w & 1) * 64 + ni * 16 + lo;
        if (o < O) {
          float val = acc[mi][ni][v] + bias[o];
          if (relu) val = fmaxf(val, 0.f);
          Out[(long long)o * 32768 + p] = val;
        }
      }
}

// ---------------------------------------------------------------------------
// Fused edge MLP (3->64->128->128) + ReLU + max over K=32.
// Block: 256 thr (8 waves), 128 positions = 4 points x 32 neighbors.
// ---------------------------------------------------------------------------
__global__ __launch_bounds__(256) void edge_mlp_max(
    const float* __restrict__ W1, const float* __restrict__ b1,   // [64,3],[64]
    const float* __restrict__ W2, const float* __restrict__ b2,   // [128,64]
    const float* __restrict__ W3, const float* __restrict__ b3,   // [128,128]
    const float* __restrict__ fch,                                // [3][262144]
    float* __restrict__ lfOut, int rowOffset)                     // [(off+c)*8192 + pt]
{
  __shared__ __bf16 act1[128][72];     // [pos][64]
  __shared__ __bf16 act2[128][136];    // [pos][128]
  __shared__ float  partial[8][128];
  __shared__ float  w1s[256];          // W1 (192) + b1 (64)
  const int tid = threadIdx.x, blk = blockIdx.x;
  const int P0 = blk * 128;
  w1s[tid] = (tid < 192) ? W1[tid] : b1[tid - 192];
  __syncthreads();

  {  // layer 1 (K=3, VALU): 2 threads per position, 32 channels each
    int pos = tid >> 1, half = tid & 1;
    float f0 = fch[0 * 262144 + P0 + pos];
    float f1 = fch[1 * 262144 + P0 + pos];
    float f2 = fch[2 * 262144 + P0 + pos];
#pragma unroll
    for (int cc = 0; cc < 32; ++cc) {
      int ch = half * 32 + cc;
      float v = w1s[ch * 3 + 0] * f0 + w1s[ch * 3 + 1] * f1 + w1s[ch * 3 + 2] * f2
              + w1s[192 + ch];
      act1[pos][ch] = f2bf(fmaxf(v, 0.f));
    }
  }
  __syncthreads();

  const int w = tid >> 5, lane = tid & 31, lo = lane & 15, hi = lane >> 4;

  {  // layer 2: [128 x 64] x [64 x 128pos], wave w owns N-tile w (16 pos)
    v8f acc[8];
#pragma unroll
    for (int mo = 0; mo < 8; ++mo) acc[mo] = v8f_zero();
#pragma unroll
    for (int kc = 0; kc < 2; ++kc) {
      v16bf bf = lds_frag_b(&act1[w * 16 + lo][0], kc * 32 + hi * 8);
#pragma unroll
      for (int mo = 0; mo < 8; ++mo) {
        v16bf af = wfrag_global(W2 + (mo * 16 + lo) * 64 + kc * 32 + hi * 8);
        acc[mo] = __builtin_amdgcn_wmma_f32_16x16x32_bf16(
            false, af, false, bf, (short)0, acc[mo], false, false);
      }
    }
#pragma unroll
    for (int mo = 0; mo < 8; ++mo)
#pragma unroll
      for (int v = 0; v < 8; ++v) {
        int o = mo * 16 + v + hi * 8;
        act2[w * 16 + lo][o] = f2bf(fmaxf(acc[mo][v] + b2[o], 0.f));
      }
  }
  __syncthreads();

  {  // layer 3: [128 x 128] x [128 x 128pos], then max over positions
    v8f acc[8];
#pragma unroll
    for (int mo = 0; mo < 8; ++mo) acc[mo] = v8f_zero();
#pragma unroll
    for (int kc = 0; kc < 4; ++kc) {
      v16bf bf = lds_frag_b(&act2[w * 16 + lo][0], kc * 32 + hi * 8);
#pragma unroll
      for (int mo = 0; mo < 8; ++mo) {
        v16bf af = wfrag_global(W3 + (mo * 16 + lo) * 128 + kc * 32 + hi * 8);
        acc[mo] = __builtin_amdgcn_wmma_f32_16x16x32_bf16(
            false, af, false, bf, (short)0, acc[mo], false, false);
      }
    }
#pragma unroll
    for (int mo = 0; mo < 8; ++mo)
#pragma unroll
      for (int v = 0; v < 8; ++v) {
        int o = mo * 16 + v + hi * 8;
        float val = acc[mo][v] + b3[o];
#pragma unroll
        for (int m = 1; m < 16; m <<= 1) val = fmaxf(val, __shfl_xor(val, m, 32));
        if (lo == 0) partial[w][o] = val;   // wave w = point w/2, k-half w%2
      }
  }
  __syncthreads();

  for (int i = tid; i < 512; i += 256) {   // 4 points x 128 channels
    int pt = i >> 7, c = i & 127;
    float m = fmaxf(partial[2 * pt][c], partial[2 * pt + 1][c]);
    lfOut[(long long)(rowOffset + c) * 8192 + (blk * 4 + pt)] = m;
  }
}

// ---------------------------------------------------------------------------
__global__ void pack_points(const float* __restrict__ coarse,
                            const float* __restrict__ org, float* __restrict__ pts) {
  int i = blockIdx.x * 256 + threadIdx.x;
  if (i >= 8 * 3 * 3072) return;
  int b = i / (3 * 3072), r = i % (3 * 3072), c = r / 3072, n = r % 3072;
  pts[i] = (n < 1024) ? coarse[(b * 3 + c) * 1024 + n]
                      : org[(b * 3 + c) * 2048 + (n - 1024)];
}

// ---------------------------------------------------------------------------
// Farthest point sampling: 1 block/batch, 1024 threads (32 waves).
// Wave-level shfl argmax + one cross-wave step: 2 barriers / iteration.
// ---------------------------------------------------------------------------
__global__ __launch_bounds__(1024) void fps_kernel(const float* __restrict__ pts,
                                                   float* __restrict__ sp) {
  __shared__ float px[3][3072];
  __shared__ float dist[3072];
  __shared__ float rv[32];
  __shared__ int   ri[32];
  __shared__ int   selS;
  const int b = blockIdx.x, tid = threadIdx.x;
  const int w = tid >> 5, lane = tid & 31;
  for (int i = tid; i < 3 * 3072; i += 1024) px[i / 3072][i % 3072] = pts[b * 3 * 3072 + i];
  for (int i = tid; i < 3072; i += 1024) dist[i] = 1e10f;
  __syncthreads();
  int sel = 0;
  if (tid < 3) sp[(b * 3 + tid) * 1024 + 0] = px[tid][0];
  for (int it = 1; it < 1024; ++it) {
    float lx = px[0][sel], ly = px[1][sel], lz = px[2][sel];
    float best = -1.f; int bestn = 0;
    for (int n = tid; n < 3072; n += 1024) {
      float dx = px[0][n] - lx, dy = px[1][n] - ly, dz = px[2][n] - lz;
      float d = fminf(dist[n], dx * dx + dy * dy + dz * dz);
      dist[n] = d;
      if (d > best) { best = d; bestn = n; }
    }
#pragma unroll
    for (int m = 1; m < 32; m <<= 1) {        // wave argmax, low-index ties
      float v2 = __shfl_xor(best, m, 32);
      int   i2 = __shfl_xor(bestn, m, 32);
      if (v2 > best || (v2 == best && i2 < bestn)) { best = v2; bestn = i2; }
    }
    if (lane == 0) { rv[w] = best; ri[w] = bestn; }
    __syncthreads();
    if (w == 0) {
      float v = rv[lane]; int id = ri[lane];
#pragma unroll
      for (int m = 1; m < 32; m <<= 1) {
        float v2 = __shfl_xor(v, m, 32);
        int   i2 = __shfl_xor(id, m, 32);
        if (v2 > v || (v2 == v && i2 < id)) { v = v2; id = i2; }
      }
      if (lane == 0) selS = id;
    }
    __syncthreads();
    sel = selS;
    if (tid < 3) sp[(b * 3 + tid) * 1024 + it] = px[tid][sel];
  }
}

// ---------------------------------------------------------------------------
// Sorted 64-NN per point (ascending d2, low index ties). 128 thr = 4 waves.
// ---------------------------------------------------------------------------
__global__ __launch_bounds__(128) void knn64(const float* __restrict__ sp,
                                             int* __restrict__ idx64) {
  __shared__ float d2[1024];
  __shared__ float rv[4];
  __shared__ int   ri[4];
  const int g = blockIdx.x, b = g >> 10, n = g & 1023, tid = threadIdx.x;
  const int w = tid >> 5, lane = tid & 31;
  const float qx = sp[(b * 3 + 0) * 1024 + n];
  const float qy = sp[(b * 3 + 1) * 1024 + n];
  const float qz = sp[(b * 3 + 2) * 1024 + n];
  for (int j = tid; j < 1024; j += 128) {
    float dx = sp[(b * 3 + 0) * 1024 + j] - qx;
    float dy = sp[(b * 3 + 1) * 1024 + j] - qy;
    float dz = sp[(b * 3 + 2) * 1024 + j] - qz;
    d2[j] = dx * dx + dy * dy + dz * dz;
  }
  __syncthreads();
  for (int r = 0; r < 64; ++r) {
    float best = 3.4e38f; int bestj = 0;
    for (int j = tid; j < 1024; j += 128) {
      float d = d2[j];
      if (d < best) { best = d; bestj = j; }
    }
#pragma unroll
    for (int m = 1; m < 32; m <<= 1) {        // wave argmin, low-index ties
      float v2 = __shfl_xor(best, m, 32);
      int   i2 = __shfl_xor(bestj, m, 32);
      if (v2 < best || (v2 == best && i2 < bestj)) { best = v2; bestj = i2; }
    }
    if (lane == 0) { rv[w] = best; ri[w] = bestj; }
    __syncthreads();
    if (tid == 0) {
      float v = rv[0]; int id = ri[0];
#pragma unroll
      for (int q = 1; q < 4; ++q)
        if (rv[q] < v || (rv[q] == v && ri[q] < id)) { v = rv[q]; id = ri[q]; }
      idx64[(long long)g * 64 + r] = id;
      d2[id] = 3.4e38f;
    }
    __syncthreads();
  }
}

// ---------------------------------------------------------------------------
// Local reference frames + rotation-invariant projection. 1 wave per point.
// ---------------------------------------------------------------------------
__global__ __launch_bounds__(256) void frame_proj(const float* __restrict__ sp,
                                                  const int* __restrict__ idx64,
                                                  float* __restrict__ fch,
                                                  float* __restrict__ fdch) {
  const int tid = threadIdx.x, w = tid >> 5, k = tid & 31;
  const int g = blockIdx.x * 8 + w, b = g >> 10, n = g & 1023;
  const float px = sp[(b * 3 + 0) * 1024 + n];
  const float py = sp[(b * 3 + 1) * 1024 + n];
  const float pz = sp[(b * 3 + 2) * 1024 + n];
  int jd = idx64[(long long)g * 64 + 2 * k];         // dilated rank 2k
  float rdx = sp[(b * 3 + 0) * 1024 + jd] - px;
  float rdy = sp[(b * 3 + 1) * 1024 + jd] - py;
  float rdz = sp[(b * 3 + 2) * 1024 + jd] - pz;
  float mx = rdx, my = rdy, mz = rdz;
#pragma unroll
  for (int m = 1; m < 32; m <<= 1) {
    mx += __shfl_xor(mx, m, 32);
    my += __shfl_xor(my, m, 32);
    mz += __shfl_xor(mz, m, 32);
  }
  mx *= (1.f / 32.f); my *= (1.f / 32.f); mz *= (1.f / 32.f);
  float pn = sqrtf(px * px + py * py + pz * pz) + 1e-8f;
  float a1x = px / pn, a1y = py / pn, a1z = pz / pn;
  float dm = mx * a1x + my * a1y + mz * a1z;
  mx -= dm * a1x; my -= dm * a1y; mz -= dm * a1z;
  float mn = sqrtf(mx * mx + my * my + mz * mz) + 1e-8f;
  float a2x = mx / mn, a2y = my / mn, a2z = mz / mn;
  float a3x = a1y * a2z - a1z * a2y;
  float a3y = a1z * a2x - a1x * a2z;
  float a3z = a1x * a2y - a1y * a2x;
  long long base = (long long)g * 32 + k;
  fdch[0 * 262144 + base] = rdx * a1x + rdy * a1y + rdz * a1z;
  fdch[1 * 262144 + base] = rdx * a2x + rdy * a2y + rdz * a2z;
  fdch[2 * 262144 + base] = rdx * a3x + rdy * a3y + rdz * a3z;
  int jk = idx64[(long long)g * 64 + k];             // plain rank k
  float rx = sp[(b * 3 + 0) * 1024 + jk] - px;
  float ry = sp[(b * 3 + 1) * 1024 + jk] - py;
  float rz = sp[(b * 3 + 2) * 1024 + jk] - pz;
  fch[0 * 262144 + base] = rx * a1x + ry * a1y + rz * a1z;
  fch[1 * 262144 + base] = rx * a2x + ry * a2y + rz * a2z;
  fch[2 * 262144 + base] = rx * a3x + ry * a3y + rz * a3z;
}

// fine = conv3_out + center (sampled point repeated SCALE times)
__global__ void add_center(const float* __restrict__ c3, const float* __restrict__ sp,
                           float* __restrict__ out) {
  int i = blockIdx.x * 256 + threadIdx.x;
  if (i >= 98304) return;
  int b = i / 12288, r = i % 12288, ch = r / 4096, j = r & 4095;
  out[24576 + i] = c3[ch * 32768 + b * 4096 + j] + sp[(b * 3 + ch) * 1024 + (j >> 2)];
}

// ---------------------------------------------------------------------------
extern "C" void kernel_launch(void* const* d_in, const int* in_sizes, int n_in,
                              void* d_out, int out_size, void* d_ws, size_t ws_size,
                              hipStream_t stream) {
  const float* x   = (const float*)d_in[0];
  const float* org = (const float*)d_in[1];
  // params flattened in sorted-dict-key pytree order:
  // convs, fc1_b, fc1_w, fc2_b, fc2_w, fc3_b, fc3_w, merge, mlp1, mlp2
  const float* cv0w = (const float*)d_in[2];   const float* cv0b = (const float*)d_in[3];
  const float* cv1w = (const float*)d_in[4];   const float* cv1b = (const float*)d_in[5];
  const float* cv2w = (const float*)d_in[6];   const float* cv2b = (const float*)d_in[7];
  const float* fc1b = (const float*)d_in[8];   const float* fc1w = (const float*)d_in[9];
  const float* fc2b = (const float*)d_in[10];  const float* fc2w = (const float*)d_in[11];
  const float* fc3b = (const float*)d_in[12];  const float* fc3w = (const float*)d_in[13];
  const float* mg0w = (const float*)d_in[14];  const float* mg0b = (const float*)d_in[15];
  const float* mg1w = (const float*)d_in[16];  const float* mg1b = (const float*)d_in[17];
  const float* mg2w = (const float*)d_in[18];  const float* mg2b = (const float*)d_in[19];
  const float* m10w = (const float*)d_in[20];  const float* m10b = (const float*)d_in[21];
  const float* m11w = (const float*)d_in[22];  const float* m11b = (const float*)d_in[23];
  const float* m12w = (const float*)d_in[24];  const float* m12b = (const float*)d_in[25];
  const float* m20w = (const float*)d_in[26];  const float* m20b = (const float*)d_in[27];
  const float* m21w = (const float*)d_in[28];  const float* m21b = (const float*)d_in[29];
  const float* m22w = (const float*)d_in[30];  const float* m22b = (const float*)d_in[31];

  float* out = (float*)d_out;                       // coarse[24576] | fine[98304]

  // workspace layout (f32 unless noted)
  float* h1    = (float*)d_ws;                      // [8][1024] row-major
  float* h2    = h1 + 8 * 1024;                     // [8][1024]
  float* pts   = h2 + 8 * 1024;                     // [8][3][3072]
  float* sp    = pts + 8 * 3 * 3072;                // [8][3][1024]
  int*   idx   = (int*)(sp + 8 * 3 * 1024);         // [8*1024][64]
  float* fch   = (float*)(idx + 8 * 1024 * 64);     // [3][262144]
  float* fdch  = fch + 3 * 262144;                  // [3][262144]
  float* lf256 = fdch + 3 * 262144;                 // [256][8192]
  float* m1    = lf256 + 256 * 8192;                // [512][8192]
  float* m2    = m1 + (size_t)512 * 8192;           // [1024][8192]
  float* lf    = m2 + (size_t)1024 * 8192;          // [1024][8192]
  float* c1    = lf + (size_t)1024 * 8192;          // [512][32768]
  float* c2    = c1 + (size_t)512 * 32768;          // [512][32768]
  float* c3    = c2 + (size_t)512 * 32768;          // [3][32768]

  dim3 blk(256);
  // fc stack (row-major activations [8][1024]):
  gemm_rm<<<dim3(1, 8),  blk, 0, stream>>>(fc1w, fc1b, x,  1024, h1, 1, 1024, 1024, 1024, 8, 1);
  gemm_rm<<<dim3(1, 8),  blk, 0, stream>>>(fc2w, fc2b, h1, 1024, h2, 1, 1024, 1024, 1024, 8, 1);
  gemm_rm<<<dim3(1, 24), blk, 0, stream>>>(fc3w, fc3b, h2, 1024, out, 1, 3072, 3072, 1024, 8, 0);

  pack_points<<<(8 * 3 * 3072 + 255) / 256, blk, 0, stream>>>(out, org, pts);
  fps_kernel<<<8, 1024, 0, stream>>>(pts, sp);
  knn64<<<8192, 128, 0, stream>>>(sp, idx);
  frame_proj<<<1024, blk, 0, stream>>>(sp, idx, fch, fdch);

  // fused edge MLPs (+ max over K) -> lf256 rows [0..127] and [128..255]
  edge_mlp_max<<<2048, blk, 0, stream>>>(m10w, m10b, m11w, m11b, m12w, m12b, fch,  lf256, 0);
  edge_mlp_max<<<2048, blk, 0, stream>>>(m20w, m20b, m21w, m21b, m22w, m22b, fdch, lf256, 128);

  // merge MLP 256 -> 512 -> 1024 -> 1024 over 8192 points (channel-major)
  gemm_cm<<<dim3(64, 4), blk, 0, stream>>>(mg0w, mg0b, lf256, m1, 512, 256, 8192, 1);
  gemm_cm<<<dim3(64, 8), blk, 0, stream>>>(mg1w, mg1b, m1,    m2, 1024, 512, 8192, 1);
  gemm_cm<<<dim3(64, 8), blk, 0, stream>>>(mg2w, mg2b, m2,    lf, 1024, 1024, 8192, 0);

  // final convs: feat (2053 ch) gathered on the fly -> 512 -> 512 -> 3 (+center)
  gemm_featgather<<<dim3(256, 4), blk, 0, stream>>>(cv0w, cv0b, x, sp, lf, c1, 512, 1);
  gemm_cm<<<dim3(256, 4), blk, 0, stream>>>(cv1w, cv1b, c1, c2, 512, 512, 32768, 1);
  gemm_cm<<<dim3(256, 1), blk, 0, stream>>>(cv2w, cv2b, c2, c3, 3, 512, 32768, 0);
  add_center<<<(98304 + 255) / 256, blk, 0, stream>>>(c3, sp, out);
}
